// ResNet_cifar100_31559419691176
// MI455X (gfx1250) — 8-bit matrix operand order repaired and checked by reading, compiled, not run on hardware
//
#include <hip/hip_runtime.h>
#include <math.h>


typedef int v8i __attribute__((ext_vector_type(8)));

__device__ __forceinline__ int quant16(float x) {
  float v = rintf(x * 4096.0f);
  v = fminf(fmaxf(v, -32768.0f), 32767.0f);
  return (int)v;
}

__global__ void quantw(const float* __restrict__ w, signed char* __restrict__ hi,
                       unsigned char* __restrict__ lo, int Cout, int K, int Kpad) {
  int i = blockIdx.x * blockDim.x + threadIdx.x;
  if (i >= Cout * Kpad) return;
  int row = i / Kpad, k = i - row * Kpad;
  int q = 0;
  if (k < K) q = quant16(w[row * K + k]);
  hi[i] = (signed char)(q >> 8);
  lo[i] = (unsigned char)(q & 255);
}

__global__ void quant_input(const float* __restrict__ x, short* __restrict__ oq,
                            int B, int C, int HW) {
  int i = blockIdx.x * blockDim.x + threadIdx.x;
  if (i >= B * C * HW) return;
  int b = i / (C * HW);
  int r = i - b * C * HW;
  int c = r / HW;
  int s = r - c * HW;
  oq[(size_t)c * B * HW + (size_t)b * HW + s] = (short)quant16(x[i]);
}

template <int KS>
__global__ void __launch_bounds__(128)
conv_wmma_iu8(const short* __restrict__ actq,
              const signed char* __restrict__ Ahi,
              const unsigned char* __restrict__ Alo,
              const float* __restrict__ bg, const float* __restrict__ bb,
              const float* __restrict__ bm, const float* __restrict__ bv,
              const float* __restrict__ res,
              float* __restrict__ out, short* __restrict__ outq,
              int Cin, int H, int W, int ohwSh, int owSh,
              int stride, int pad, int Kpad, int Bn, int relu) {
  const int lane = threadIdx.x;
  const int m0   = threadIdx.y * 16;
  const int n0   = blockIdx.x * 16;
  const int HW = H * W;
  const int Nin = Bn * HW, Nout = Bn << ohwSh;
  const int K = Cin * KS * KS;

  __shared__ unsigned int sBh[256];
  __shared__ unsigned int sBl[256];
  __shared__ int sPB[16], sPY[16], sPX[16];
  unsigned char* Bh = (unsigned char*)sBh;
  unsigned char* Bl = (unsigned char*)sBl;

  const int T = blockDim.y * 32;
  const int perT = 1024 / T;
  const int tid = threadIdx.y * 32 + lane;

  if (tid < 16) {
    int n = n0 + tid;
    int b = n >> ohwSh;
    int rem = n & ((1 << ohwSh) - 1);
    int oy = rem >> owSh, ox = rem & ((1 << owSh) - 1);
    sPB[tid] = b * HW;
    sPY[tid] = oy * stride - pad;
    sPX[tid] = ox * stride - pad;
  }
  __syncthreads();

  v8i ahh = {}, amid = {}, all_ = {};

  for (int c0 = 0; c0 < Kpad; c0 += 64) {
    for (int i = 0; i < perT; ++i) {
      int idx = tid * perT + i;
      int p = idx >> 6, k = idx & 63;
      int kk = c0 + k;
      int q = 0;
      if (kk < K) {
        int cin = kk / (KS * KS);
        int rr = kk - cin * (KS * KS);
        int ky = rr / KS, kx = rr - ky * KS;
        int iy = sPY[p] + ky;
        int ix = sPX[p] + kx;
        if ((unsigned)iy < (unsigned)H && (unsigned)ix < (unsigned)W)
          q = (int)actq[(size_t)cin * Nin + sPB[p] + iy * W + ix];
      }
      Bh[idx] = (unsigned char)((q >> 8) & 0xFF);
      Bl[idx] = (unsigned char)(q & 0xFF);
    }
    __syncthreads();

    v8i bh, bl;
    {
      const int col = lane & 15;
      const int kb = (lane < 16) ? 0 : 16;
#pragma unroll
      for (int v = 0; v < 8; ++v) {
        int ko = ((v >> 2) * 32) + kb + ((v & 3) * 4);
        int o = (col * 64 + ko) >> 2;
        bh[v] = (int)sBh[o];
        bl[v] = (int)sBl[o];
      }
    }

    v8i ah, al;
    {
      const int row = m0 + (lane & 15);
      const int kofs = (lane < 16) ? 0 : 16;
      const signed char*   pa = Ahi + (size_t)row * Kpad + c0;
      const unsigned char* pb = Alo + (size_t)row * Kpad + c0;
#pragma unroll
      for (int v = 0; v < 8; ++v) {
        int ko = ((v >> 2) * 32) + ((v & 3) * 4) + kofs;
        ah[v] = *(const int*)(pa + ko);
        al[v] = *(const int*)(pb + ko);
      }
    }

    ahh  = __builtin_amdgcn_wmma_i32_16x16x64_iu8(true,  ah, true,  bh, ahh,  false, false);
    amid = __builtin_amdgcn_wmma_i32_16x16x64_iu8(true,  ah, false, bl, amid, false, false);
    amid = __builtin_amdgcn_wmma_i32_16x16x64_iu8(false, al, true,  bh, amid, false, false);
    all_ = __builtin_amdgcn_wmma_i32_16x16x64_iu8(false, al, false, bl, all_, false, false);
    __syncthreads();
  }

  const int col = lane & 15;
  const int pix = n0 + col;
  const int mofs = (lane < 16) ? 0 : 8;
#pragma unroll
  for (int r = 0; r < 8; ++r) {
    int chan = m0 + r + mofs;
    long long v = ((long long)ahh[r] << 16)
                + ((long long)amid[r] << 8)
                + (long long)all_[r];
    if (v >  2147483647LL) v =  2147483647LL;
    if (v < -2147483648LL) v = -2147483648LL;
    float y = (float)(int)v * (1.0f / 16777216.0f);
    float sc = bg[chan] * rsqrtf(bv[chan] + 1e-5f);
    y = sc * (y - bm[chan]) + bb[chan];
    if (res) y += res[(size_t)chan * Nout + pix];
    if (relu) y = fmaxf(y, 0.0f);
    out[(size_t)chan * Nout + pix] = y;
    if (outq) outq[(size_t)chan * Nout + pix] = (short)quant16(y);
  }
}

__global__ void avgpool8(const float* __restrict__ act, float* __restrict__ pooled, int B) {
  int i = blockIdx.x * blockDim.x + threadIdx.x;
  if (i >= B * 64) return;
  int b = i >> 6, c = i & 63;
  const float* p = act + (size_t)c * B * 64 + (size_t)b * 64;
  float s = 0.0f;
  for (int k = 0; k < 64; ++k) s += p[k];
  pooled[i] = s * (1.0f / 64.0f);
}

__global__ void head_kernel(const float* __restrict__ pooled,
                            const float* __restrict__ g20, const float* __restrict__ b20,
                            const float* __restrict__ m20, const float* __restrict__ v20,
                            const float* __restrict__ fcw, const float* __restrict__ fcb,
                            const float* __restrict__ g21, const float* __restrict__ b21,
                            const float* __restrict__ m21, const float* __restrict__ v21,
                            float* __restrict__ out) {
  __shared__ float hv[64];
  __shared__ float red[128];
  int b = blockIdx.x, t = threadIdx.x;
  if (t < 64) {
    float p = pooled[b * 64 + t];
    hv[t] = g20[t] * (p - m20[t]) * rsqrtf(v20[t] + 1e-5f) + b20[t];
  }
  __syncthreads();
  float logit = -3.402823466e38f;
  if (t < 100) {
    float s = fcb[t];
#pragma unroll
    for (int c = 0; c < 64; ++c) s += hv[c] * fcw[t * 64 + c];
    logit = g21[t] * (s - m21[t]) * rsqrtf(v21[t] + 1e-5f) + b21[t];
  }
  red[t] = logit;
  __syncthreads();
  for (int o = 64; o > 0; o >>= 1) {
    if (t < o) red[t] = fmaxf(red[t], red[t + o]);
    __syncthreads();
  }
  float mx = red[0];
  __syncthreads();
  red[t] = (t < 100) ? expf(logit - mx) : 0.0f;
  __syncthreads();
  for (int o = 64; o > 0; o >>= 1) {
    if (t < o) red[t] += red[t + o];
    __syncthreads();
  }
  float lse = mx + logf(red[0]);
  if (t < 100) out[b * 100 + t] = logit - lse;
}

static inline int ilog2i(int v) { int s = 0; while ((1 << s) < v) ++s; return s; }

extern "C" void kernel_launch(void* const* d_in, const int* in_sizes, int n_in,
                              void* d_out, int out_size, void* d_ws, size_t ws_size,
                              hipStream_t stream) {
  const float* x    = (const float*)d_in[0];
  const float* w1   = (const float*)d_in[1];
  const float* ws16 = (const float*)d_in[2];
  const float* w8   = (const float*)d_in[3];
  const float* wr1  = (const float*)d_in[4];
  const float* ws32 = (const float*)d_in[5];
  const float* w14  = (const float*)d_in[6];
  const float* wr2  = (const float*)d_in[7];
  const float* ws64 = (const float*)d_in[8];
  const float* fcw  = (const float*)d_in[9];
  const float* fcb  = (const float*)d_in[10];
  const float* g16  = (const float*)d_in[11];
  const float* b16  = (const float*)d_in[12];
  const float* m16  = (const float*)d_in[13];
  const float* v16  = (const float*)d_in[14];
  const float* g32  = (const float*)d_in[15];
  const float* b32  = (const float*)d_in[16];
  const float* m32  = (const float*)d_in[17];
  const float* v32  = (const float*)d_in[18];
  const float* g64  = (const float*)d_in[19];
  const float* b64  = (const float*)d_in[20];
  const float* m64  = (const float*)d_in[21];
  const float* v64  = (const float*)d_in[22];
  const float* g20  = (const float*)d_in[23];
  const float* b20  = (const float*)d_in[24];
  const float* m20  = (const float*)d_in[25];
  const float* v20  = (const float*)d_in[26];
  const float* g21  = (const float*)d_in[27];
  const float* b21  = (const float*)d_in[28];
  const float* m21  = (const float*)d_in[29];
  const float* v21  = (const float*)d_in[30];
  float* out = (float*)d_out;

  char* ws = (char*)d_ws;
  size_t off = 0;
  auto alloc = [&](size_t bytes) -> char* {
    char* p = ws + off;
    off += (bytes + 255) & ~(size_t)255;
    return p;
  };
  const int B = 512;
  const size_t ACT_BYTES = (size_t)16 * B * 1024 * sizeof(float);
  const size_t QACT_BYTES = (size_t)16 * B * 1024 * sizeof(short);
  float* bufA = (float*)alloc(ACT_BYTES);
  float* bufB = (float*)alloc(ACT_BYTES);
  float* bufC = (float*)alloc(ACT_BYTES);
  short* qA = (short*)alloc(QACT_BYTES);
  short* qB = (short*)alloc(QACT_BYTES);
  float* pooled = (float*)alloc((size_t)B * 64 * sizeof(float));

  struct QW { signed char* hi; unsigned char* lo; int Cout, K, Kpad; };
  auto mkq = [&](const float* w, int Cout, int K) -> QW {
    QW q;
    q.Cout = Cout; q.K = K; q.Kpad = (K + 63) & ~63;
    q.hi = (signed char*)alloc((size_t)Cout * q.Kpad);
    q.lo = (unsigned char*)alloc((size_t)Cout * q.Kpad);
    int n = Cout * q.Kpad;
    quantw<<<(n + 255) / 256, 256, 0, stream>>>(w, q.hi, q.lo, Cout, K, q.Kpad);
    return q;
  };

  QW qc1 = mkq(w1, 16, 27);
  QW q16c[6]; for (int l = 0; l < 6; ++l) q16c[l] = mkq(ws16 + (size_t)l * 16 * 144, 16, 144);
  QW qw8  = mkq(w8, 32, 144);
  QW qr1  = mkq(wr1, 32, 16);
  QW q32c[5]; for (int l = 0; l < 5; ++l) q32c[l] = mkq(ws32 + (size_t)l * 32 * 288, 32, 288);
  QW qw14 = mkq(w14, 64, 288);
  QW qr2  = mkq(wr2, 64, 32);
  QW q64c[5]; for (int l = 0; l < 5; ++l) q64c[l] = mkq(ws64 + (size_t)l * 64 * 576, 64, 576);

  auto conv = [&](const short* inq, const QW& q,
                  const float* g, const float* bbp, const float* mmp, const float* vvp,
                  const float* res, float* o, short* oq,
                  int Cin, int H, int W, int stride, int pad, int ks, int relu) {
    int OH = (H + 2 * pad - ks) / stride + 1;
    int OW = (W + 2 * pad - ks) / stride + 1;
    int Nout = B * OH * OW;
    int ohwSh = ilog2i(OH * OW);
    int owSh = ilog2i(OW);
    dim3 grid(Nout / 16, 1, 1);
    dim3 blk(32, q.Cout / 16, 1);
    if (ks == 3)
      conv_wmma_iu8<3><<<grid, blk, 0, stream>>>(inq, q.hi, q.lo, g, bbp, mmp, vvp,
                                                 res, o, oq, Cin, H, W, ohwSh, owSh,
                                                 stride, pad, q.Kpad, B, relu);
    else
      conv_wmma_iu8<1><<<grid, blk, 0, stream>>>(inq, q.hi, q.lo, g, bbp, mmp, vvp,
                                                 res, o, oq, Cin, H, W, ohwSh, owSh,
                                                 stride, pad, q.Kpad, B, relu);
  };

  {
    int n = B * 3 * 1024;
    quant_input<<<(n + 255) / 256, 256, 0, stream>>>(x, qA, B, 3, 1024);
  }
  conv(qA, qc1, g16, b16, m16, v16, nullptr, bufA, qB, 3, 32, 32, 1, 1, 3, 1);

  float* h = bufA; float* t1 = bufC; float* t2 = bufB;
  for (int i = 0; i < 3; ++i) {
    conv(qB, q16c[2 * i],     g16 + (1 + 2 * i) * 16, b16 + (1 + 2 * i) * 16,
                              m16 + (1 + 2 * i) * 16, v16 + (1 + 2 * i) * 16,
         nullptr, t1, qA, 16, 32, 32, 1, 1, 3, 1);
    conv(qA, q16c[2 * i + 1], g16 + (2 + 2 * i) * 16, b16 + (2 + 2 * i) * 16,
                              m16 + (2 + 2 * i) * 16, v16 + (2 + 2 * i) * 16,
         h, t2, qB, 16, 32, 32, 1, 1, 3, 1);
    float* tmp = h; h = t2; t2 = tmp;
  }
  conv(qB, qw8, g32, b32, m32, v32, nullptr, t1, qA, 16, 32, 32, 2, 1, 3, 1);
  conv(qB, qr1, g32 + 6 * 32, b32 + 6 * 32, m32 + 6 * 32, v32 + 6 * 32,
       nullptr, t2, nullptr, 16, 32, 32, 2, 0, 1, 0);
  conv(qA, q32c[0], g32 + 1 * 32, b32 + 1 * 32, m32 + 1 * 32, v32 + 1 * 32,
       t2, h, qB, 32, 16, 16, 1, 1, 3, 1);
  for (int i = 0; i < 2; ++i) {
    conv(qB, q32c[1 + 2 * i], g32 + (2 + 2 * i) * 32, b32 + (2 + 2 * i) * 32,
                              m32 + (2 + 2 * i) * 32, v32 + (2 + 2 * i) * 32,
         nullptr, t1, qA, 32, 16, 16, 1, 1, 3, 1);
    conv(qA, q32c[2 + 2 * i], g32 + (3 + 2 * i) * 32, b32 + (3 + 2 * i) * 32,
                              m32 + (3 + 2 * i) * 32, v32 + (3 + 2 * i) * 32,
         h, t2, qB, 32, 16, 16, 1, 1, 3, 1);
    float* tmp = h; h = t2; t2 = tmp;
  }
  conv(qB, qw14, g64, b64, m64, v64, nullptr, t1, qA, 32, 16, 16, 2, 1, 3, 1);
  conv(qB, qr2, g64 + 6 * 64, b64 + 6 * 64, m64 + 6 * 64, v64 + 6 * 64,
       nullptr, t2, nullptr, 32, 16, 16, 2, 0, 1, 0);
  conv(qA, q64c[0], g64 + 1 * 64, b64 + 1 * 64, m64 + 1 * 64, v64 + 1 * 64,
       t2, h, qB, 64, 8, 8, 1, 1, 3, 1);
  for (int i = 0; i < 2; ++i) {
    conv(qB, q64c[1 + 2 * i], g64 + (2 + 2 * i) * 64, b64 + (2 + 2 * i) * 64,
                              m64 + (2 + 2 * i) * 64, v64 + (2 + 2 * i) * 64,
         nullptr, t1, qA, 64, 8, 8, 1, 1, 3, 1);
    conv(qA, q64c[2 + 2 * i], g64 + (3 + 2 * i) * 64, b64 + (3 + 2 * i) * 64,
                              m64 + (3 + 2 * i) * 64, v64 + (3 + 2 * i) * 64,
         h, t2, qB, 64, 8, 8, 1, 1, 3, 1);
    float* tmp = h; h = t2; t2 = tmp;
  }
  avgpool8<<<(B * 64 + 255) / 256, 256, 0, stream>>>(h, pooled, B);
  head_kernel<<<B, 128, 0, stream>>>(pooled, g20, b20, m20, v20, fcw, fcb,
                                     g21, b21, m21, v21, out);

  (void)in_sizes; (void)n_in; (void)out_size; (void)ws_size;
}
